// OldNoiseConv_34943853920608
// MI455X (gfx1250) — matrix operand order repaired and checked by reading, not compiled, not run on hardware
//
#include <hip/hip_runtime.h>

// ---------------------------------------------------------------------------
// NoiseConv for MI455X (gfx1250, wave32).
//
// y[o,p] = sum_i W[o,i]*(1 + 0.1*eps[p,o,i])*xf[i,p] + b[o],  out[o*P+p].
//
// Roofline: eps = 512 MB f32 streamed exactly once -> ~22 us floor at
// 23.3 TB/s; useful FLOPs ~0.27 G -> purely bandwidth bound. The WMMA diag
// mapping keeps operand prep at ~1 FMA + 1 cvt per eps element:
//   A'(ob,kb)[M,k] = xf[i,p]*(1+0.1*eps[p,ob*16+M,i])   (per-point, bf16)
//   B(ob,kb)[k,n]  = W[ob*16+n,i]                       (stationary, bf16)
//   diag(A'B)[t]   = y[ob*16+t,p] - b[.]
// 8x v_wmma_f32_16x16x32_bf16 per point, C-chained over the K=64 split.
// The 16x matrix-FLOP waste of taking only diag(D) is free vs the HBM floor.
// Diagonal extraction is branch-free (v_cndmask tree on lane&7), one
// predicated store per o-block.
// ---------------------------------------------------------------------------

typedef __attribute__((ext_vector_type(16))) __bf16 v16bf;
typedef __attribute__((ext_vector_type(8)))  float  v8f;

constexpr int   CI    = 64;
constexpr int   CO    = 64;
constexpr int   P     = 32768;        // B*NS*NPTS (power of two)
constexpr int   BLOCK = 256;          // 8 wave32 waves
constexpr int   GRID  = 1024;         // 8192 waves -> 4 points per wave
constexpr float NOISE = 0.1f;

__global__ __launch_bounds__(BLOCK)
void noiseconv_wmma(const float* __restrict__ x,    // viewed as [CI][P]
                    const float* __restrict__ W,    // [CO][CI]
                    const float* __restrict__ bias, // [CO]
                    const float* __restrict__ eps,  // [P][CO][CI]
                    float* __restrict__ out)        // [CO][P]
{
  const int lane = threadIdx.x & 31;
  const int half = lane >> 4;                       // 16-lane group
  const int m    = lane & 15;                       // tile row (A) / col (B/C)
  const int wid  = blockIdx.x * (BLOCK / 32) + (threadIdx.x >> 5);
  const int nw   = GRID * (BLOCK / 32);

  // Diagonal lanes: D[t][t] lives at lane t (t=0..7) and lane 24+r (t=8+r).
  const bool diagAct = (lane < 8) || (lane >= 24);

  // --- Stationary B operands (built once per wave, reused for all points).
  // wave32 bf16 B layout (matches A): lanes 0-15 own K={0-7,16-23}, lanes
  // 16-31 own K={8-15,24-31}. With B[k][n]=W[ob*16+n][kb*32+k] this lane
  // reads two 8-float runs of one W row (L0-resident, 16 KB total).
  v16bf Bop[2][4];
  #pragma unroll
  for (int kb = 0; kb < 2; ++kb) {
    #pragma unroll
    for (int ob = 0; ob < 4; ++ob) {
      const float4* w4 =
          (const float4*)(W + (ob * 16 + m) * CI + kb * 32 + half * 8);
      #pragma unroll
      for (int q = 0; q < 4; ++q) {
        float4 v = w4[q + (q & 2)];
        Bop[kb][ob][4 * q + 0] = (__bf16)v.x;
        Bop[kb][ob][4 * q + 1] = (__bf16)v.y;
        Bop[kb][ob][4 * q + 2] = (__bf16)v.z;
        Bop[kb][ob][4 * q + 3] = (__bf16)v.w;
      }
    }
  }

  float bv[4];
  #pragma unroll
  for (int ob = 0; ob < 4; ++ob) bv[ob] = bias[ob * 16 + m];

  #pragma unroll 1                                  // keep one code copy
  for (int p = wid; p < P; p += nw) {
    const float* ep = eps + (size_t)p * (CO * CI);

    // Unconditional speculative prefetch of the next point's 16 KB eps tile
    // (global_prefetch_b8); wrap the index instead of branching.
    {
      const float* np = eps + (size_t)((p + nw) & (P - 1)) * (CO * CI);
      __builtin_prefetch(np + lane * 128, 0, 0);       // 512B stride / lane
      __builtin_prefetch(np + lane * 128 + 64, 0, 0);  // +256B
    }

    v8f c[4];
    #pragma unroll
    for (int ob = 0; ob < 4; ++ob)
      #pragma unroll
      for (int r = 0; r < 8; ++r) c[ob][r] = 0.0f;

    #pragma unroll
    for (int kb = 0; kb < 2; ++kb) {
      // x values for the 16 K slots this lane owns in the A layout:
      // lane<16 owns K={0-7,16-23}, lane>=16 owns K={8-15,24-31} (+kb*32).
      // Addresses are uniform within each 16-lane group; x is L2-resident.
      const int i0 = kb * 32 + half * 8;
      float xv[16], sx[16];
      #pragma unroll
      for (int j = 0; j < 8; ++j) {
        xv[j]     = x[(i0 + j)      * P + p];
        xv[j + 8] = x[(i0 + 16 + j) * P + p];
      }
      #pragma unroll
      for (int j = 0; j < 16; ++j) sx[j] = NOISE * xv[j];

      #pragma unroll
      for (int ob = 0; ob < 4; ++ob) {
        // This lane's slice of eps row (ob*16+m): two 8-float runs, 16B
        // aligned -> 4x global_load_b128; every eps byte loaded exactly once.
        const float4* e4 =
            (const float4*)(ep + (ob * 16 + m) * CI + kb * 32 + half * 8);
        float4 e0 = e4[0], e1 = e4[1], e2 = e4[4], e3 = e4[5];
        float e[16] = {e0.x, e0.y, e0.z, e0.w, e1.x, e1.y, e1.z, e1.w,
                       e2.x, e2.y, e2.z, e2.w, e3.x, e3.y, e3.z, e3.w};

        v16bf a;
        #pragma unroll
        for (int j = 0; j < 16; ++j)
          a[j] = (__bf16)__builtin_fmaf(e[j], sx[j], xv[j]); // x*(1+0.1*eps)

        // D = A' x B + C  (16x16x32 bf16 -> f32); EXEC is all-ones here
        // (all control flow up to this point is wave-uniform).
        c[ob] = __builtin_amdgcn_wmma_f32_16x16x32_bf16(
            false, a, false, Bop[kb][ob], (short)0, c[ob], false, false);
      }
    }

    // Branch-free diagonal extraction. C layout: VGPR r = D[r][lane]
    // (lanes 0-15) / D[8+r][lane-16] (lanes 16-31). A cndmask tree on
    // lane&7 puts cc[lane&7] in every lane, which equals D[t][t] at the
    // diagonal lanes (0-7 and 24-31). One predicated store per o-block;
    // o = ob*16 + (lane&15) for both lane groups.
    #pragma unroll
    for (int ob = 0; ob < 4; ++ob) {
      v8f cc = c[ob];
      float s0 = (lane & 1) ? cc[1] : cc[0];
      float s1 = (lane & 1) ? cc[3] : cc[2];
      float s2 = (lane & 1) ? cc[5] : cc[4];
      float s3 = (lane & 1) ? cc[7] : cc[6];
      float t0 = (lane & 2) ? s1 : s0;
      float t1 = (lane & 2) ? s3 : s2;
      float dv = (lane & 4) ? t1 : t0;
      if (diagAct) out[(size_t)(ob * 16 + m) * P + p] = dv + bv[ob];
    }
  }
}

extern "C" void kernel_launch(void* const* d_in, const int* in_sizes, int n_in,
                              void* d_out, int out_size, void* d_ws,
                              size_t ws_size, hipStream_t stream) {
  const float* x   = (const float*)d_in[0];
  const float* W   = (const float*)d_in[1];
  const float* b   = (const float*)d_in[2];
  const float* eps = (const float*)d_in[3];
  float*       out = (float*)d_out;
  noiseconv_wmma<<<GRID, BLOCK, 0, stream>>>(x, W, b, eps, out);
  (void)in_sizes; (void)n_in; (void)out_size; (void)d_ws; (void)ws_size;
}
